// GraphTransformerLayer_21766894256448
// MI455X (gfx1250) — matrix operand order repaired and checked by reading, not compiled, not run on hardware
//
#include <hip/hip_runtime.h>

typedef __attribute__((ext_vector_type(16))) _Float16 v16h;
typedef __attribute__((ext_vector_type(8)))  _Float16 h8v;
typedef __attribute__((ext_vector_type(4)))  _Float16 h4v;
typedef __attribute__((ext_vector_type(8)))  float    v8f;
typedef __attribute__((ext_vector_type(4)))  float    f4v;

#define NN   100000
#define EE   1600000
#define HIDD 128
#define NHEAD 4
#define HD   32

// -------- gfx1250 async global->LDS path (guarded; falls back to ds_store) ----
#if defined(__has_builtin)
#if __has_builtin(__builtin_amdgcn_global_load_async_to_lds_b128)
#define HAVE_ASYNC_LDS 1
#endif
#endif
#ifndef HAVE_ASYNC_LDS
#define HAVE_ASYNC_LDS 0
#endif

// builtin signature (from hipcc diagnostic): (AS1 v4i*, AS3 v4i*, imm int, imm int)
typedef int v4i_t __attribute__((vector_size(16)));
typedef __attribute__((address_space(1))) v4i_t gv4i_t;
typedef __attribute__((address_space(3))) v4i_t lv4i_t;

__device__ __forceinline__ void wait_async_all() {
#if HAVE_ASYNC_LDS
#if __has_builtin(__builtin_amdgcn_s_wait_asynccnt)
  __builtin_amdgcn_s_wait_asynccnt(0);
#else
  asm volatile("s_wait_asynccnt 0" ::: "memory");
#endif
#endif
}

// ---------------- LayerNorm (one wave32 per row of 128) ----------------
__global__ __launch_bounds__(128) void ln_kernel(const float* __restrict__ x,
                                                 const float* __restrict__ g,
                                                 const float* __restrict__ b,
                                                 _Float16* __restrict__ y, int nrows) {
  int wave = threadIdx.x >> 5, lane = threadIdx.x & 31;
  int row = blockIdx.x * 4 + wave;
  if (row >= nrows) return;
  const float* xr = x + (size_t)row * HIDD;
  f4v v = *(const f4v*)(xr + lane * 4);
  float s = v[0] + v[1] + v[2] + v[3];
  #pragma unroll
  for (int off = 16; off > 0; off >>= 1) s += __shfl_xor(s, off, 32);
  float mean = s * (1.0f / HIDD);
  f4v d = v - mean;
  float sq = d[0]*d[0] + d[1]*d[1] + d[2]*d[2] + d[3]*d[3];
  #pragma unroll
  for (int off = 16; off > 0; off >>= 1) sq += __shfl_xor(sq, off, 32);
  float rstd = rsqrtf(sq * (1.0f / HIDD) + 1e-5f);
  f4v gg = *(const f4v*)(g + lane * 4);
  f4v bb = *(const f4v*)(b + lane * 4);
  f4v o = d * rstd * gg + bb;
  h4v ho; ho[0] = (_Float16)o[0]; ho[1] = (_Float16)o[1];
  ho[2] = (_Float16)o[2]; ho[3] = (_Float16)o[3];
  *(h4v*)(y + (size_t)row * HIDD + lane * 4) = ho;
}

// ---------------- Weight conversion / packing to f16 ----------------
__global__ __launch_bounds__(256) void convert_weights(
    const float* __restrict__ Wq, const float* __restrict__ Wk, const float* __restrict__ Wv,
    const float* __restrict__ bq, const float* __restrict__ bk, const float* __restrict__ bv,
    const float* __restrict__ Wo, const float* __restrict__ W1, const float* __restrict__ W2,
    _Float16* __restrict__ Wqkv_h, _Float16* __restrict__ Wo_h,
    _Float16* __restrict__ W1_h, _Float16* __restrict__ W2_h, float* __restrict__ bqkv) {
  int i = blockIdx.x * 256 + threadIdx.x;
  if (i < HIDD * 384) {   // packed [k][384] : Q | K | V columns
    int k = i / 384, n = i % 384;
    float w = (n < 128) ? Wq[k * 128 + n] : (n < 256 ? Wk[k * 128 + (n - 128)]
                                                     : Wv[k * 128 + (n - 256)]);
    Wqkv_h[i] = (_Float16)w;
  }
  if (i < HIDD * HIDD) Wo_h[i] = (_Float16)Wo[i];
  if (i < HIDD * 2 * HIDD) { W1_h[i] = (_Float16)W1[i]; W2_h[i] = (_Float16)W2[i]; }
  if (i < 384) bqkv[i] = (i < 128) ? bq[i] : (i < 256 ? bk[i - 128] : bv[i - 256]);
}

// ---------------- WMMA GEMM with LDS-staged B ----------------
// Block = 256 threads = 8 waves; wave w owns M-tile (blockIdx.x*8+w)*16.
// B is consumed in 64-column chunks staged cooperatively into LDS via
// global_load_async_to_lds_b128 (fallback: load + ds_store). LDS row stride
// padded to 72 halfs to break lane-per-K ds_load bank aliasing.
// A row-major [M][KDIM] (f16, or f32 converted on load); B f16 row-major [KDIM][ldb].
template <int KDIM, bool A_F32, bool RELU, bool RES, bool OUT16>
__global__ __launch_bounds__(256) void gemm_wmma(const void* __restrict__ Aptr, int lda,
                                                 const _Float16* __restrict__ B, int ldb,
                                                 const float* __restrict__ bias,
                                                 const float* __restrict__ resid,
                                                 void* __restrict__ outp, int ldo,
                                                 int Nout, int Mrows) {
  __shared__ __align__(16) _Float16 sB[KDIM * 72];   // 64 data + 8 pad halfs per K-row

  const int tid  = threadIdx.x;
  const int wave = tid >> 5, lane = tid & 31;
  const int r = lane & 15, hi = lane >> 4;

  const int m0 = (blockIdx.x * 8 + wave) * 16;
  const bool valid = (m0 < Mrows);          // wave-uniform; no early return (barriers!)
  const int mbase = valid ? m0 : 0;         // clamp loads for tail waves

  const int nch = Nout >> 6;                // 64-column chunks
  for (int c = 0; c < nch; ++c) {
    const int n0 = c * 64;

    // ---- cooperative stage of B[K x 64] chunk into LDS ----
    #pragma unroll
    for (int j = 0; j < KDIM / 32; ++j) {
      int cid = j * 256 + tid;              // 16-byte segment id
      int k = cid >> 3, seg = cid & 7;
      const _Float16* gp = B + (size_t)k * ldb + n0 + seg * 8;
      _Float16* lp = &sB[k * 72 + seg * 8];
#if HAVE_ASYNC_LDS
      __builtin_amdgcn_global_load_async_to_lds_b128((gv4i_t*)gp, (lv4i_t*)lp, 0, 0);
#else
      *(h8v*)lp = *(const h8v*)gp;
#endif
    }
    wait_async_all();
    __syncthreads();                        // chunk visible to all 8 waves

    // ---- compute 16x64 strip for this wave's M-tile ----
    v8f acc[4];
    #pragma unroll
    for (int t = 0; t < 4; ++t) acc[t] = (v8f){};

    for (int kb = 0; kb < KDIM / 32; ++kb) {
      const int kbase = kb * 32;
      v16h a;
      if (A_F32) {
        const float* A = (const float*)Aptr + (size_t)(mbase + r) * lda + kbase + hi * 8;
        v8f x0 = *(const v8f*)A;
        v8f x1 = *(const v8f*)(A + 16);
        #pragma unroll
        for (int j = 0; j < 8; ++j) { a[j] = (_Float16)x0[j]; a[8 + j] = (_Float16)x1[j]; }
      } else {
        const _Float16* A = (const _Float16*)Aptr + (size_t)(mbase + r) * lda + kbase + hi * 8;
        h8v a0 = *(const h8v*)A;
        h8v a1 = *(const h8v*)(A + 16);
        #pragma unroll
        for (int j = 0; j < 8; ++j) { a[j] = a0[j]; a[8 + j] = a1[j]; }
      }
      const _Float16* bp = &sB[(kbase + hi * 8) * 72 + r];   // lane == N column; K rows hi*8+j, 16+hi*8+j
      #pragma unroll
      for (int t = 0; t < 4; ++t) {
        v16h bf;
        #pragma unroll
        for (int j = 0; j < 8; ++j) { bf[j] = bp[j * 72 + t * 16]; bf[8 + j] = bp[(16 + j) * 72 + t * 16]; }
        acc[t] = __builtin_amdgcn_wmma_f32_16x16x32_f16(false, a, false, bf,
                                                        (short)0, acc[t], false, false);
      }
    }

    // ---- epilogue for this chunk ----
    if (valid) {
      #pragma unroll
      for (int t = 0; t < 4; ++t) {
        int col = n0 + t * 16 + r;
        float bb = bias[col];
        #pragma unroll
        for (int v = 0; v < 8; ++v) {
          int row = m0 + v + (hi ? 8 : 0);
          float val = acc[t][v] + bb;
          if (RES) val += resid[(size_t)row * ldo + col];
          if (RELU) val = fmaxf(val, 0.0f);
          if (OUT16) ((_Float16*)outp)[(size_t)row * ldo + col] = (_Float16)val;
          else       ((float*)outp)[(size_t)row * ldo + col] = val;
        }
      }
    }
    __syncthreads();                        // all reads done before next chunk overwrites sB
  }
}

// ---------------- Edge phase ----------------
__device__ __forceinline__ unsigned enc_f32(float f) {
  unsigned u = __float_as_uint(f);
  return (u & 0x80000000u) ? ~u : (u | 0x80000000u);
}
__device__ __forceinline__ float dec_f32(unsigned e) {
  unsigned u = (e & 0x80000000u) ? (e & 0x7FFFFFFFu) : ~e;
  return __uint_as_float(u);
}

__global__ __launch_bounds__(256) void edge_scores(const int* __restrict__ src,
                                                   const int* __restrict__ dst,
                                                   const _Float16* __restrict__ QKV,
                                                   float* __restrict__ scores,
                                                   unsigned* __restrict__ nmax) {
  int t = blockIdx.x * 256 + threadIdx.x;
  if (t >= EE * NHEAD) return;
  int e = t >> 2, hh = t & 3;
  int si = src[e], di = dst[e];
  const _Float16* qp = QKV + (size_t)di * 384 + hh * HD;         // Q cols 0..127
  const _Float16* kp = QKV + (size_t)si * 384 + 128 + hh * HD;   // K cols 128..255
  float acc = 0.f;
  #pragma unroll
  for (int c = 0; c < 4; ++c) {
    h8v q = *(const h8v*)(qp + c * 8);
    h8v k = *(const h8v*)(kp + c * 8);
    #pragma unroll
    for (int j = 0; j < 8; ++j) acc += (float)q[j] * (float)k[j];
  }
  acc *= 0.17677669529663687f;  // 1/sqrt(32)
  scores[t] = acc;
  atomicMax(nmax + (size_t)di * NHEAD + hh, enc_f32(acc));
}

__global__ __launch_bounds__(256) void edge_expsum(const int* __restrict__ dst,
                                                   float* __restrict__ scores,
                                                   const unsigned* __restrict__ nmax,
                                                   float* __restrict__ denom) {
  int t = blockIdx.x * 256 + threadIdx.x;
  if (t >= EE * NHEAD) return;
  int e = t >> 2, hh = t & 3;
  int di = dst[e];
  float m = dec_f32(nmax[(size_t)di * NHEAD + hh]);
  float ex = __expf(scores[t] - m);
  scores[t] = ex;
  unsafeAtomicAdd(denom + (size_t)di * NHEAD + hh, ex);
}

__global__ __launch_bounds__(256) void edge_aggregate(const int* __restrict__ src,
                                                      const int* __restrict__ dst,
                                                      const _Float16* __restrict__ QKV,
                                                      const float* __restrict__ scores,
                                                      const float* __restrict__ denom,
                                                      float* __restrict__ hnew) {
  int t = blockIdx.x * 256 + threadIdx.x;
  if (t >= EE * NHEAD) return;
  int e = t >> 2, hh = t & 3;
  int si = src[e], di = dst[e];
  float alpha = scores[t] / denom[(size_t)di * NHEAD + hh];
  const _Float16* vp = QKV + (size_t)si * 384 + 256 + hh * HD;   // V cols 256..383
  float* op = hnew + (size_t)di * HIDD + hh * HD;
  #pragma unroll
  for (int c = 0; c < 4; ++c) {
    h8v vv = *(const h8v*)(vp + c * 8);
    #pragma unroll
    for (int j = 0; j < 8; ++j) unsafeAtomicAdd(op + c * 8 + j, alpha * (float)vv[j]);
  }
}

// ---------------- Host orchestration ----------------
extern "C" void kernel_launch(void* const* d_in, const int* in_sizes, int n_in,
                              void* d_out, int out_size, void* d_ws, size_t ws_size,
                              hipStream_t stream) {
  const float* h   = (const float*)d_in[0];
  const int*   src = (const int*)d_in[1];
  const int*   dst = (const int*)d_in[2];
  const float* Wq  = (const float*)d_in[3];
  const float* bq  = (const float*)d_in[4];
  const float* Wk  = (const float*)d_in[5];
  const float* bk  = (const float*)d_in[6];
  const float* Wv  = (const float*)d_in[7];
  const float* bv  = (const float*)d_in[8];
  const float* Wo  = (const float*)d_in[9];
  const float* bo  = (const float*)d_in[10];
  const float* W1  = (const float*)d_in[11];
  const float* b1  = (const float*)d_in[12];
  const float* W2  = (const float*)d_in[13];
  const float* b2  = (const float*)d_in[14];
  const float* g1  = (const float*)d_in[15];
  const float* be1 = (const float*)d_in[16];
  const float* g2  = (const float*)d_in[17];
  const float* be2 = (const float*)d_in[18];

  char* ws = (char*)d_ws;
  size_t off = 0;
  auto alloc = [&](size_t bytes) -> void* {
    void* p = ws + off;
    off = (off + bytes + 255) & ~(size_t)255;
    return p;
  };
  _Float16* hn     = (_Float16*)alloc((size_t)NN * HIDD * 2);   // LN1 out, reused for LN2 out
  _Float16* QKV    = (_Float16*)alloc((size_t)NN * 384 * 2);    // Q|K|V f16, reused for FFN1 out
  float*    sc     = (float*)alloc((size_t)EE * NHEAD * 4);     // scores -> exp(scores)
  unsigned* nmax   = (unsigned*)alloc((size_t)NN * NHEAD * 4);
  float*    den    = (float*)alloc((size_t)NN * NHEAD * 4);
  float*    hnew   = (float*)alloc((size_t)NN * HIDD * 4);
  float*    hres   = (float*)alloc((size_t)NN * HIDD * 4);
  _Float16* Wqkv_h = (_Float16*)alloc((size_t)HIDD * 384 * 2);
  _Float16* Wo_h   = (_Float16*)alloc((size_t)HIDD * HIDD * 2);
  _Float16* W1_h   = (_Float16*)alloc((size_t)HIDD * 2 * HIDD * 2);
  _Float16* W2_h   = (_Float16*)alloc((size_t)2 * HIDD * HIDD * 2);
  float*    bqkv   = (float*)alloc(384 * 4);

  (void)hipMemsetAsync(nmax, 0, (size_t)NN * NHEAD * 4, stream);  // encoded 0 < any real
  (void)hipMemsetAsync(den,  0, (size_t)NN * NHEAD * 4, stream);
  (void)hipMemsetAsync(hnew, 0, (size_t)NN * HIDD * 4, stream);

  convert_weights<<<192, 256, 0, stream>>>(Wq, Wk, Wv, bq, bk, bv, Wo, W1, W2,
                                           Wqkv_h, Wo_h, W1_h, W2_h, bqkv);

  ln_kernel<<<(NN + 3) / 4, 128, 0, stream>>>(h, g1, be1, hn, NN);

  const int mtiles  = NN / 16;              // 6250
  const int mblocks = (mtiles + 7) / 8;     // 782 (last block: 2 active waves)

  // QKV = LN1(h) @ [Wq|Wk|Wv] + [bq|bk|bv]   (f16 out)
  gemm_wmma<128, false, false, false, true><<<mblocks, 256, 0, stream>>>(
      hn, HIDD, Wqkv_h, 384, bqkv, nullptr, QKV, 384, 384, NN);

  int egrid = (EE * NHEAD + 255) / 256;
  edge_scores<<<egrid, 256, 0, stream>>>(src, dst, QKV, sc, nmax);
  edge_expsum<<<egrid, 256, 0, stream>>>(dst, sc, nmax, den);
  edge_aggregate<<<egrid, 256, 0, stream>>>(src, dst, QKV, sc, den, hnew);

  // hres = h + hnew @ Wo + bo   (f32 A converted on load, f32 out)
  gemm_wmma<128, true, false, true, false><<<mblocks, 256, 0, stream>>>(
      hnew, HIDD, Wo_h, HIDD, bo, h, hres, HIDD, 128, NN);

  ln_kernel<<<(NN + 3) / 4, 128, 0, stream>>>(hres, g2, be2, hn, NN);

  // f1 = relu(LN2 @ W1 + b1)   (f16 out, reuses QKV buffer)
  gemm_wmma<128, false, true, false, true><<<mblocks, 256, 0, stream>>>(
      hn, HIDD, W1_h, 256, b1, nullptr, QKV, 256, 256, NN);

  // out = hres + f1 @ W2 + b2  (f32 out)
  gemm_wmma<256, false, false, true, false><<<mblocks, 256, 0, stream>>>(
      QKV, 256, W2_h, HIDD, b2, hres, d_out, HIDD, 128, NN);
}
